// Head_68719476736254
// MI455X (gfx1250) — compile-verified, matrix operand order checked by reading, not run on hardware
//
#include <hip/hip_runtime.h>
#include <hip/hip_bf16.h>
#include <cmath>

// ---------------------------------------------------------------------------
// Single-head causal attention, flash-attention style, bf16 WMMA (gfx1250).
//   x:  [B=4, T=4096, C=1024] f32
//   Wq/Wk/Wv: [1024, 64] f32
//   out: [B, T, 64] f32
// K/V tiles (64 keys) double-buffered into LDS via the Tensor Data Mover.
// ---------------------------------------------------------------------------

#define N_EMBD 1024
#define HEAD   64
#define BATCH  4
#define TLEN   4096
// reference scale C**-0.5 (C=1024 -> 1/32), folded with log2(e) for exp2 softmax
#define QSCALE (0.03125f * 1.44269504088896f)

typedef __attribute__((ext_vector_type(16))) __bf16 v16bf;
typedef __attribute__((ext_vector_type(8)))  float  v8f;
typedef __attribute__((ext_vector_type(4)))  unsigned v4u;
typedef __attribute__((ext_vector_type(8)))  int      v8i;
typedef __attribute__((ext_vector_type(4)))  int      v4i;

union Frag16 {              // one WMMA 16-bit A/B operand (16 bf16 = 8 VGPRs)
    v16bf v;
    unsigned short u[16];
    unsigned w[8];
    uint4 q[2];
};
union H8 {                  // 8 packed bf16
    uint4 q;
    unsigned w[4];
    unsigned short u[8];
};

__device__ __forceinline__ unsigned short f2bf(float f) {
    union { float f; unsigned u; } x; x.f = f;
    unsigned r = x.u + 0x7FFFu + ((x.u >> 16) & 1u);   // round-to-nearest-even
    return (unsigned short)(r >> 16);
}

#if __has_builtin(__builtin_amdgcn_cvt_pk_bf16_f32)
typedef __attribute__((ext_vector_type(2))) __bf16 v2bf;
__device__ __forceinline__ unsigned pk2bf(float x, float y) {
    union { v2bf v; unsigned u; } c;
    c.v = __builtin_amdgcn_cvt_pk_bf16_f32(x, y);      // v_cvt_pk_bf16_f32
    return c.u;
}
#else
__device__ __forceinline__ unsigned pk2bf(float x, float y) {
    return (unsigned)f2bf(x) | ((unsigned)f2bf(y) << 16);
}
#endif

__device__ __forceinline__ v8f wmma_bf16(const Frag16& a, const Frag16& b, v8f c) {
    // (neg_a, A, neg_b, B, c_mod, C, reuse_a, reuse_b)
    return __builtin_amdgcn_wmma_f32_16x16x32_bf16(false, a.v, false, b.v,
                                                   (short)0, c, false, false);
}

// ---------------------------------------------------------------------------
// TDM: 2-D tile load, global (bf16 elements) -> LDS.  D# per ISA §8.3/8.4:
//   group0: count=1 | lds_addr | global_addr | type=2
//   group1: data_size=2B, tensor/tile dims, dim0 stride
// 6-arg builtin (clang-23 / therock-10.0 lane).  Completion: s_wait_tensorcnt.
// ---------------------------------------------------------------------------
#if __has_builtin(__builtin_amdgcn_tensor_load_to_lds)
#define HAVE_TDM 1
__device__ __forceinline__ void tdm_load_2d(unsigned lds_addr, const void* gptr,
                                            unsigned tile_d0, unsigned tile_d1,
                                            unsigned stride0) {
    const unsigned long long ga = (unsigned long long)(uintptr_t)gptr;
    v4u g0;
    g0[0] = 1u;                                        // count=1 (valid), user D#
    g0[1] = lds_addr;                                  // LDS byte address
    g0[2] = (unsigned)(ga & 0xFFFFFFFFu);              // global_addr[31:0]
    g0[3] = (unsigned)((ga >> 32) & 0x01FFFFFFu)       // global_addr[56:32]
          | (2u << 30);                                // type=2 ("image")
    v8i g1;
    g1[0] = (int)(1u << 16);                           // wg_mask=0, data_size=1 (2B)
    g1[1] = (int)((tile_d0 & 0xFFFFu) << 16);          // tensor_dim0 = tile_d0
    g1[2] = (int)((tile_d1 & 0xFFFFu) << 16);          // tensor_dim1 = tile_d1
    g1[3] = (int)((tile_d0 & 0xFFFFu) << 16);          // tile_dim0
    g1[4] = (int)(tile_d1 & 0xFFFFu);                  // tile_dim1 (tile_dim2=0)
    g1[5] = (int)stride0;                              // tensor_dim0_stride[31:0]
    g1[6] = 0;                                         // stride0[47:32], stride1 lo
    g1[7] = 0;                                         // stride1 hi (unused, 2-D)
    const v4i gz4 = {0, 0, 0, 0};
    const v8i gz8 = {0, 0, 0, 0, 0, 0, 0, 0};
    __builtin_amdgcn_tensor_load_to_lds(g0, g1, gz4, gz4, gz8, 0);
}
#endif

// ---------------------------------------------------------------------------
// Kernel 0: W[C][H] f32  ->  Wt[3][H][C] bf16 (transposed, contiguous in C)
// ---------------------------------------------------------------------------
__global__ void __launch_bounds__(256)
wt_convert(const float* __restrict__ Wq, const float* __restrict__ Wk,
           const float* __restrict__ Wv, unsigned short* __restrict__ Wt) {
    int idx = blockIdx.x * 256 + threadIdx.x;          // 0 .. 3*65536-1
    int m = idx >> 16;
    int e = idx & 65535;
    int h = e >> 10;                                   // 0..63
    int c = e & 1023;                                  // 0..1023
    const float* W = (m == 0) ? Wq : (m == 1) ? Wk : Wv;
    Wt[(size_t)m * 65536 + h * 1024 + c] = f2bf(W[c * 64 + h]);
}

// ---------------------------------------------------------------------------
// Kernel 1: QKV projection.  Per block: 128 rows of x; per wave: 16 rows.
// Q (pre-scaled by QSCALE), K stored [B][T][64] bf16; V transposed [B][64][T].
// ---------------------------------------------------------------------------
__global__ void __launch_bounds__(256)
qkv_proj(const float* __restrict__ x, const unsigned short* __restrict__ Wt,
         unsigned short* __restrict__ Q, unsigned short* __restrict__ K,
         unsigned short* __restrict__ Vt) {
    const int wave  = threadIdx.x >> 5;
    const int lane  = threadIdx.x & 31;
    const int b     = blockIdx.x >> 5;                 // 32 row-blocks per batch
    const int tblk  = blockIdx.x & 31;
    const int row0  = tblk * 128 + wave * 16;
    const int mrow  = lane & 15;
    const int koff  = (lane >> 4) << 3;                // 0 or 8 (A/B lane layout)
    const int half8 = koff;                            // C-layout row offset

    const float* xrow = x + ((size_t)b * TLEN + row0 + mrow) * N_EMBD;

    v8f aq[4], ak[4], av[4];
    #pragma unroll
    for (int n = 0; n < 4; ++n)
        #pragma unroll
        for (int j = 0; j < 8; ++j) { aq[n][j] = 0.f; ak[n][j] = 0.f; av[n][j] = 0.f; }

    for (int kk = 0; kk < N_EMBD; kk += 32) {
        __builtin_prefetch(xrow + kk + 128, 0, 1);     // global_prefetch_b8

        // ---- A fragment: x rows, f32 -> packed bf16 ------------------------
        Frag16 a;
        const float* p0 = xrow + kk + koff;
        const float* p1 = xrow + kk + 16 + koff;
        float4 f0 = *(const float4*)(p0);
        float4 f1 = *(const float4*)(p0 + 4);
        float4 f2 = *(const float4*)(p1);
        float4 f3 = *(const float4*)(p1 + 4);
        a.w[0] = pk2bf(f0.x, f0.y);  a.w[1] = pk2bf(f0.z, f0.w);
        a.w[2] = pk2bf(f1.x, f1.y);  a.w[3] = pk2bf(f1.z, f1.w);
        a.w[4] = pk2bf(f2.x, f2.y);  a.w[5] = pk2bf(f2.z, f2.w);
        a.w[6] = pk2bf(f3.x, f3.y);  a.w[7] = pk2bf(f3.z, f3.w);

        // ---- B fragments from Wt (contiguous per lane) ---------------------
        const unsigned short* wbase = Wt + mrow * 1024 + kk + koff;
        #pragma unroll
        for (int n = 0; n < 4; ++n) {
            const unsigned short* pq = wbase + 0 * 65536 + n * 16 * 1024;
            const unsigned short* pk = wbase + 1 * 65536 + n * 16 * 1024;
            const unsigned short* pv = wbase + 2 * 65536 + n * 16 * 1024;
            Frag16 bq, bk, bv;
            bq.q[0] = *(const uint4*)(pq);  bq.q[1] = *(const uint4*)(pq + 16);
            bk.q[0] = *(const uint4*)(pk);  bk.q[1] = *(const uint4*)(pk + 16);
            bv.q[0] = *(const uint4*)(pv);  bv.q[1] = *(const uint4*)(pv + 16);
            aq[n] = wmma_bf16(a, bq, aq[n]);
            ak[n] = wmma_bf16(a, bk, ak[n]);
            av[n] = wmma_bf16(a, bv, av[n]);
        }
    }

    // ---- Store: Q (pre-scaled), K scatter b16; Vt packed contiguous --------
    #pragma unroll
    for (int n = 0; n < 4; ++n) {
        const int h = n * 16 + mrow;
        #pragma unroll
        for (int j = 0; j < 8; ++j) {
            const size_t ridx = ((size_t)b * TLEN + row0 + j + half8) * HEAD + h;
            Q[ridx] = f2bf(aq[n][j] * QSCALE);
            K[ridx] = f2bf(ak[n][j]);
        }
        H8 pack;
        #pragma unroll
        for (int j = 0; j < 4; ++j) pack.w[j] = pk2bf(av[n][2 * j], av[n][2 * j + 1]);
        *(uint4*)(Vt + ((size_t)b * HEAD + h) * TLEN + row0 + half8) = pack.q;
    }
}

// ---------------------------------------------------------------------------
// Kernel 2: flash attention (causal, online softmax, base-2 domain).
// Block = 128 query rows (8 waves x 16).  64-key tiles double-buffered in LDS
// by the TDM: tile i+1's DMA runs under tile i's WMMAs.
// ---------------------------------------------------------------------------
__global__ void __launch_bounds__(256)
flash_attn(const unsigned short* __restrict__ Q, const unsigned short* __restrict__ K,
           const unsigned short* __restrict__ Vt, float* __restrict__ out) {
    __shared__ unsigned short kbuf[2][64 * 64];        // K tile  [s][h], 8KB each
    __shared__ unsigned short vbuf[2][64 * 64];        // Vt tile [h][s], 8KB each
    __shared__ unsigned short pbuf[8][16 * 64];        // per-wave P transpose buf

    const int wave  = threadIdx.x >> 5;
    const int lane  = threadIdx.x & 31;
    const int b     = blockIdx.x >> 5;
    const int tblk  = blockIdx.x & 31;
    const int row0  = tblk * 128 + wave * 16;
    const int mrow  = lane & 15;
    const int koff  = (lane >> 4) << 3;
    const int half8 = koff;

    // Q fragments (A operand, K=h split into 2 chunks of 32) — held in regs
    Frag16 fq[2];
    const unsigned short* qrow = Q + ((size_t)b * TLEN + row0 + mrow) * HEAD;
    #pragma unroll
    for (int kh = 0; kh < 2; ++kh) {
        fq[kh].q[0] = *(const uint4*)(qrow + kh * 32 + koff);
        fq[kh].q[1] = *(const uint4*)(qrow + kh * 32 + 16 + koff);
    }

    v8f o[4];
    float mrun[8], lrun[8];
    #pragma unroll
    for (int n = 0; n < 4; ++n)
        #pragma unroll
        for (int j = 0; j < 8; ++j) o[n][j] = 0.f;
    #pragma unroll
    for (int j = 0; j < 8; ++j) { mrun[j] = -INFINITY; lrun[j] = 0.f; }

    const int nIter = (tblk + 1) * 2;                  // 64-key tiles to process

#ifdef HAVE_TDM
    if (wave == 0) {                                   // prologue: tile 0 DMA
        tdm_load_2d((unsigned)(uintptr_t)(void*)kbuf[0],
                    K + (size_t)b * TLEN * HEAD, 64, 64, HEAD);
        tdm_load_2d((unsigned)(uintptr_t)(void*)vbuf[0],
                    Vt + (size_t)b * HEAD * TLEN, 64, 64, TLEN);
    }
#endif

    for (int it = 0; it < nIter; ++it) {
        const int s0  = it * 64;
        const int cur = it & 1;
#ifdef HAVE_TDM
        if (wave == 0) {
            if (it + 1 < nIter) {                      // prefetch next tile
                const int nxt = cur ^ 1;
                const int s1  = s0 + 64;
                tdm_load_2d((unsigned)(uintptr_t)(void*)kbuf[nxt],
                            K + ((size_t)b * TLEN + s1) * HEAD, 64, 64, HEAD);
                tdm_load_2d((unsigned)(uintptr_t)(void*)vbuf[nxt],
                            Vt + (size_t)b * HEAD * TLEN + s1, 64, 64, TLEN);
                __builtin_amdgcn_s_wait_tensorcnt(2);  // tile `it` complete
            } else {
                __builtin_amdgcn_s_wait_tensorcnt(0);
            }
        }
#else
        {
            const int el = threadIdx.x * 16;           // 0..4080; [64][64] tiles
            const int r8 = el >> 6, c8 = el & 63;
            *(uint4*)(kbuf[cur] + el) =
                *(const uint4*)(K + ((size_t)b * TLEN + s0 + r8) * HEAD + c8);
            *(uint4*)(kbuf[cur] + el + 8) =
                *(const uint4*)(K + ((size_t)b * TLEN + s0 + r8) * HEAD + c8 + 8);
            *(uint4*)(vbuf[cur] + el) =
                *(const uint4*)(Vt + ((size_t)b * HEAD + r8) * TLEN + s0 + c8);
            *(uint4*)(vbuf[cur] + el + 8) =
                *(const uint4*)(Vt + ((size_t)b * HEAD + r8) * TLEN + s0 + c8 + 8);
        }
#endif
        __syncthreads();

        if (s0 <= row0 + 15) {                         // wave has unmasked work
            const unsigned short* kb_ = kbuf[cur];
            const unsigned short* vb_ = vbuf[cur];

            // ---- S = Q @ K^T : four 16x16 tiles ---------------------------
            v8f sacc[4];
            #pragma unroll
            for (int nt = 0; nt < 4; ++nt)
                #pragma unroll
                for (int j = 0; j < 8; ++j) sacc[nt][j] = 0.f;
            #pragma unroll
            for (int nt = 0; nt < 4; ++nt) {
                #pragma unroll
                for (int kh = 0; kh < 2; ++kh) {
                    Frag16 bk;
                    const unsigned short* kb =
                        kb_ + (nt * 16 + mrow) * 64 + kh * 32 + koff;
                    bk.q[0] = *(const uint4*)(kb);
                    bk.q[1] = *(const uint4*)(kb + 16);
                    sacc[nt] = wmma_bf16(fq[kh], bk, sacc[nt]);
                }
            }

            // ---- causal mask (scale already folded into Q) ----------------
            float t2[4][8];
            #pragma unroll
            for (int nt = 0; nt < 4; ++nt)
                #pragma unroll
                for (int j = 0; j < 8; ++j) {
                    const int r    = row0 + j + half8;
                    const int scol = s0 + nt * 16 + mrow;
                    t2[nt][j] = (scol <= r) ? sacc[nt][j] : -INFINITY;
                }

            // ---- online softmax: row max/sum across 16-lane halves --------
            float alpha[8];
            #pragma unroll
            for (int j = 0; j < 8; ++j) {
                float v = fmaxf(fmaxf(t2[0][j], t2[1][j]),
                                fmaxf(t2[2][j], t2[3][j]));
                #pragma unroll
                for (int d = 1; d < 16; d <<= 1) v = fmaxf(v, __shfl_xor(v, d, 32));
                const float mn = fmaxf(mrun[j], v);
                alpha[j] = exp2f(mrun[j] - mn);        // 0 on first hit
                mrun[j]  = mn;
            }

            unsigned short* pb = pbuf[wave];
            float rsum[8];
            #pragma unroll
            for (int j = 0; j < 8; ++j) rsum[j] = 0.f;
            #pragma unroll
            for (int nt = 0; nt < 4; ++nt)
                #pragma unroll
                for (int j = 0; j < 8; ++j) {
                    const float p = exp2f(t2[nt][j] - mrun[j]);
                    rsum[j] += p;
                    pb[(j + half8) * 64 + nt * 16 + mrow] = f2bf(p);
                }
            #pragma unroll
            for (int j = 0; j < 8; ++j) {
                float v = rsum[j];
                #pragma unroll
                for (int d = 1; d < 16; d <<= 1) v += __shfl_xor(v, d, 32);
                lrun[j] = lrun[j] * alpha[j] + v;
            }

            // ---- rescale O, then O += P @ V (two 32-deep k-steps) ---------
            #pragma unroll
            for (int n = 0; n < 4; ++n)
                #pragma unroll
                for (int j = 0; j < 8; ++j) o[n][j] *= alpha[j];

            Frag16 fp[2];                              // P as A operand (K = s)
            #pragma unroll
            for (int ks = 0; ks < 2; ++ks) {
                const unsigned short* pr = pb + mrow * 64 + ks * 32 + koff;
                fp[ks].q[0] = *(const uint4*)(pr);
                fp[ks].q[1] = *(const uint4*)(pr + 16);
            }
            #pragma unroll
            for (int n = 0; n < 4; ++n) {
                #pragma unroll
                for (int ks = 0; ks < 2; ++ks) {
                    Frag16 bv;
                    const unsigned short* vb =
                        vb_ + (n * 16 + mrow) * 64 + ks * 32 + koff;
                    bv.q[0] = *(const uint4*)(vb);
                    bv.q[1] = *(const uint4*)(vb + 16);
                    o[n] = wmma_bf16(fp[ks], bv, o[n]);
                }
            }
        }
        __syncthreads();
    }

    // ---- epilogue: divide by row sum, store fp32 ---------------------------
    #pragma unroll
    for (int n = 0; n < 4; ++n) {
        const int h = n * 16 + mrow;
        #pragma unroll
        for (int j = 0; j < 8; ++j) {
            const int r = row0 + j + half8;
            out[((size_t)b * TLEN + r) * HEAD + h] = o[n][j] / lrun[j];
        }
    }
}

// ---------------------------------------------------------------------------
extern "C" void kernel_launch(void* const* d_in, const int* in_sizes, int n_in,
                              void* d_out, int out_size, void* d_ws, size_t ws_size,
                              hipStream_t stream) {
    (void)in_sizes; (void)n_in; (void)out_size; (void)ws_size;
    const float* x  = (const float*)d_in[0];
    const float* Wq = (const float*)d_in[1];
    const float* Wk = (const float*)d_in[2];
    const float* Wv = (const float*)d_in[3];
    float* out = (float*)d_out;

    // workspace layout (bf16 elements):
    //   Wt  : 3 * 64 * 1024           = 196608
    //   Q   : B * T * 64              = 1048576
    //   K   : B * T * 64              = 1048576
    //   Vt  : B * 64 * T              = 1048576
    unsigned short* Wt  = (unsigned short*)d_ws;
    unsigned short* Qb  = Wt + 3 * 65536;
    unsigned short* Kb  = Qb + (size_t)BATCH * TLEN * HEAD;
    unsigned short* Vtb = Kb + (size_t)BATCH * TLEN * HEAD;

    wt_convert<<<768, 256, 0, stream>>>(Wq, Wk, Wv, Wt);
    qkv_proj  <<<BATCH * (TLEN / 128), 256, 0, stream>>>(x, Wt, Qb, Kb, Vtb);
    flash_attn<<<BATCH * (TLEN / 128), 256, 0, stream>>>(Qb, Kb, Vtb, out);
}
